// MultiHeadAttention_19782619365417
// MI455X (gfx1250) — matrix operand order repaired and checked by reading, not compiled, not run on hardware
//
#include <hip/hip_runtime.h>
#include <hip/hip_bf16.h>

// ---------------------------------------------------------------------------
// MultiHeadAttention for MI455X (gfx1250, wave32, WMMA).
// prep (f32->f16, W transpose) -> fused QKV proj (WMMA, 4 token-tiles/wave)
// -> flash attention (WMMA, 32 queries/wave, no LDS, packed-shuffle P relayout)
// -> out proj (WMMA, 4 token-tiles/wave).
// ---------------------------------------------------------------------------

typedef __attribute__((ext_vector_type(16))) _Float16 v16h;
typedef __attribute__((ext_vector_type(8)))  _Float16 v8h;
typedef __attribute__((ext_vector_type(8)))  float    v8f;
typedef __attribute__((ext_vector_type(8)))  int      v8i;

#define NB   8
#define SEQ  2048
#define DIM  256
#define NH   4
#define HD   64
#define TOK  (NB * SEQ)          // 16384
#define QT   2                   // query tiles per attention wave (32 queries)
#define TB   4                   // token tiles per projection wave (64 tokens)

__device__ __forceinline__ v8f wmma_f16(v16h a, v16h b, v8f c) {
    return __builtin_amdgcn_wmma_f32_16x16x32_f16(
        false, a, false, b, (short)0, c, false, false);
}

// Pack two f32 -> packed f16 dword (v_cvt_pk_rtz_f16_f32).
__device__ __forceinline__ int pack_f16(float a, float b) {
    return __builtin_bit_cast(int, __builtin_amdgcn_cvt_pkrtz(a, b));
}

// A-operand fragment (16x32 f16, M=row per lane&15, K linear per half):
// lanes 0-15: K = cb+0..15 ; lanes 16-31: K = cb+16..31.
__device__ __forceinline__ v16h load_a_frag(const _Float16* __restrict__ row,
                                            int cb, int lane) {
    int ks = lane & 16;                           // 0 or 16
    return *(const v16h*)(row + cb + ks);
}

// B-operand fragment (32x16 f16, N=col per lane&15, K linear per half):
// lanes 0-15: K = cb+0..15 ; lanes 16-31: K = cb+16..31.
// `col` points at the row of B^T storage (column n of B), contiguous in K.
__device__ __forceinline__ v16h load_b_frag(const _Float16* __restrict__ col,
                                            int cb, int lane) {
    int ks = lane & 16;                           // 0 or 16
    return *(const v16h*)(col + cb + ks);
}

// ---------------------------------------------------------------------------
// Kernel 1: x -> f16, weights -> transposed f16 (B^T row-major).
// ---------------------------------------------------------------------------
__global__ void prep_kernel(const float* __restrict__ x,
                            const float* __restrict__ Wq, const float* __restrict__ Wk,
                            const float* __restrict__ Wv, const float* __restrict__ Wo,
                            _Float16* __restrict__ xh,
                            _Float16* __restrict__ wqt, _Float16* __restrict__ wkt,
                            _Float16* __restrict__ wvt, _Float16* __restrict__ wot) {
    int stride = gridDim.x * blockDim.x;
    int idx = blockIdx.x * blockDim.x + threadIdx.x;
    for (int i = idx; i < TOK * DIM; i += stride) xh[i] = (_Float16)x[i];
    for (int i = idx; i < DIM * DIM; i += stride) {
        int n = i >> 8, k = i & 255;              // Wt[n][k] = W[k][n]
        int src = k * DIM + n;
        wqt[i] = (_Float16)Wq[src];
        wkt[i] = (_Float16)Wk[src];
        wvt[i] = (_Float16)Wv[src];
        wot[i] = (_Float16)Wo[src];
    }
}

// ---------------------------------------------------------------------------
// Kernel 2: fused QKV projection, 4 token-tiles x 1 out-tile per wave.
// Weight fragments loaded once per k-chunk, reused by 4 token tiles.
// Q gets 0.125*log2(e) folded in (softmax runs in exp2 domain).
// Qh/Kh: [b][h][s][d] f16.  Vt: [b][h][d][s] f16 (computed as Wv^T @ x^T).
// ---------------------------------------------------------------------------
__global__ void __launch_bounds__(128, 1)
qkv_kernel(const _Float16* __restrict__ xh,
           const _Float16* __restrict__ wqt,
           const _Float16* __restrict__ wkt,
           const _Float16* __restrict__ wvt,
           const float* __restrict__ bq, const float* __restrict__ bk,
           const float* __restrict__ bv,
           _Float16* __restrict__ Qh, _Float16* __restrict__ Kh,
           _Float16* __restrict__ Vt) {
    int lane = threadIdx.x & 31;
    int wid  = threadIdx.x >> 5;
    int gid  = blockIdx.x * (blockDim.x >> 5) + wid;   // 0..4095
    int nt = gid & 15;                                 // out-dim tile
    int ts = gid >> 4;                                 // token superblock 0..255
    int t0 = ts * (16 * TB), n0 = nt * 16;
    int col = lane & 15;

    const _Float16* wqc = wqt + (n0 + col) * DIM;      // B columns (out-dims)
    const _Float16* wkc = wkt + (n0 + col) * DIM;
    const _Float16* wvr = wvt + (n0 + col) * DIM;      // A rows (out-dims) for V^T

    v8f cq[TB], ck[TB], cv[TB];
#pragma unroll
    for (int t = 0; t < TB; ++t) { cq[t] = {}; ck[t] = {}; cv[t] = {}; }

#pragma unroll
    for (int c = 0; c < 8; ++c) {
        int cb = c * 32;
        v16h qf = load_b_frag(wqc, cb, lane);
        v16h kf = load_b_frag(wkc, cb, lane);
        v16h vf = load_a_frag(wvr, cb, lane);
#pragma unroll
        for (int t = 0; t < TB; ++t) {
            const _Float16* xrow = xh + (t0 + t * 16 + col) * DIM;
            v16h xa = load_a_frag(xrow, cb, lane);
            v16h xb = load_b_frag(xrow, cb, lane);
            cq[t] = wmma_f16(xa, qf, cq[t]);
            ck[t] = wmma_f16(xa, kf, ck[t]);
            cv[t] = wmma_f16(vf, xb, cv[t]);           // V^T tile: M=out-dim, N=token
        }
    }

    int b = t0 / SEQ;                                  // 64-token block within one b
    int h = n0 >> 6;
    int d0 = n0 & 63;
    int rowoff = (lane >> 4) << 3;                     // 0 or 8
    int bhS = (b * NH + h) * SEQ;
    float bqv = bq[n0 + col];
    float bkv = bk[n0 + col];
    const float QSCALE = 0.125f * 1.4426950408889634f; // 1/sqrt(hd) * log2(e)
#pragma unroll
    for (int t = 0; t < TB; ++t) {
        int s_base = t0 - b * SEQ + t * 16;
#pragma unroll
        for (int r = 0; r < 8; ++r) {
            int s = s_base + r + rowoff;
            Qh[(bhS + s) * HD + d0 + col] = (_Float16)((cq[t][r] + bqv) * QSCALE);
            Kh[(bhS + s) * HD + d0 + col] = (_Float16)(ck[t][r] + bkv);
            int dv = d0 + r + rowoff;
            Vt[((b * NH + h) * HD + dv) * SEQ + (s_base + col)] =
                (_Float16)(cv[t][r] + bv[n0 + r + rowoff]);
        }
    }
}

// ---------------------------------------------------------------------------
// Kernel 3: flash attention, one wave per (b, h, 32-query block).
// S^T = K @ Q^T so softmax rows are per-lane; the C-layout of the two 16-key
// score tiles becomes the B-operand of O^T = V^T @ P after a packed
// shfl_xor(16) half-exchange. K/V fragments shared across 2 query tiles.
// ---------------------------------------------------------------------------
__global__ void __launch_bounds__(128, 1)
attn_kernel(const _Float16* __restrict__ Qh,
            const _Float16* __restrict__ Kh,
            const _Float16* __restrict__ Vt,
            _Float16* __restrict__ ctxh) {
    int lane = threadIdx.x & 31;
    int wid  = threadIdx.x >> 5;
    int gid  = blockIdx.x * (blockDim.x >> 5) + wid;   // 0..2047
    int bh = gid >> 6;                                 // 0..31
    int q0 = (gid & 63) * (16 * QT);
    int col = lane & 15;
    bool hihalf = lane >= 16;

    const _Float16* Qp = Qh + (bh * SEQ + q0) * HD;
    const _Float16* Kp = Kh + bh * SEQ * HD;
    const _Float16* Vp = Vt + bh * HD * SEQ;

    v16h qb[QT][2];
#pragma unroll
    for (int t = 0; t < QT; ++t) {
        qb[t][0] = load_b_frag(Qp + (t * 16 + col) * HD, 0,  lane);
        qb[t][1] = load_b_frag(Qp + (t * 16 + col) * HD, 32, lane);
    }

    v8f co[QT][4];                                     // O^T tiles per d-range
#pragma unroll
    for (int t = 0; t < QT; ++t)
#pragma unroll
        for (int d = 0; d < 4; ++d) co[t][d] = {};
    float m[QT], l[QT];
#pragma unroll
    for (int t = 0; t < QT; ++t) { m[t] = -1e30f; l[t] = 0.f; }

    for (int j = 0; j < SEQ; j += 32) {
        const _Float16* kr0 = Kp + (j + col) * HD;
        const _Float16* kr1 = Kp + (j + 16 + col) * HD;
        v16h ka00 = load_a_frag(kr0, 0, lane);
        v16h ka01 = load_a_frag(kr0, 32, lane);
        v16h ka10 = load_a_frag(kr1, 0, lane);
        v16h ka11 = load_a_frag(kr1, 32, lane);

        v16h pb[QT];
#pragma unroll
        for (int t = 0; t < QT; ++t) {
            v8f s0 = {}, s1 = {};
            s0 = wmma_f16(ka00, qb[t][0], s0);
            s0 = wmma_f16(ka01, qb[t][1], s0);
            s1 = wmma_f16(ka10, qb[t][0], s1);
            s1 = wmma_f16(ka11, qb[t][1], s1);

            // row max (query per lane; keys across 16 regs + lane halves)
            float vm = fmaxf(s0[0], s0[1]);
#pragma unroll
            for (int i = 2; i < 8; ++i) vm = fmaxf(vm, s0[i]);
#pragma unroll
            for (int i = 0; i < 8; ++i) vm = fmaxf(vm, s1[i]);
            vm = fmaxf(vm, __shfl_xor(vm, 16, 32));
            float mnew = fmaxf(m[t], vm);

            // p = exp2(s - m) (scores already in exp2 domain)
            float rs = 0.f;
#pragma unroll
            for (int i = 0; i < 8; ++i) { s0[i] = exp2f(s0[i] - mnew); rs += s0[i]; }
#pragma unroll
            for (int i = 0; i < 8; ++i) { s1[i] = exp2f(s1[i] - mnew); rs += s1[i]; }
            rs += __shfl_xor(rs, 16, 32);

            // rescale only when some lane's max moved (uniform branch: EXEC
            // stays all-ones around the WMMAs).
            if (__ballot(mnew > m[t]) != 0ull) {
                float alpha = exp2f(m[t] - mnew);
                l[t] = l[t] * alpha + rs;
#pragma unroll
                for (int d = 0; d < 4; ++d)
#pragma unroll
                    for (int i = 0; i < 8; ++i) co[t][d][i] *= alpha;
            } else {
                l[t] += rs;
            }
            m[t] = mnew;

            // Pack P to f16 pairs, half-exchange packed dwords, assemble B frag:
            // lanes 0-15 need keys j..j+15, lanes 16-31 need keys j+16..j+31.
            v8i wv;
#pragma unroll
            for (int i = 0; i < 4; ++i) {
                int p0 = pack_f16(s0[2 * i], s0[2 * i + 1]);
                int p1 = pack_f16(s1[2 * i], s1[2 * i + 1]);
                int x0 = __shfl_xor(p0, 16, 32);
                int x1 = __shfl_xor(p1, 16, 32);
                wv[i]     = hihalf ? x1 : p0;
                wv[i + 4] = hihalf ? p1 : x0;
            }
            pb[t] = __builtin_bit_cast(v16h, wv);
        }

        // O^T += V^T(tile) @ P ; V fragments shared by all query tiles.
#pragma unroll
        for (int d = 0; d < 4; ++d) {
            v16h va = load_a_frag(Vp + (d * 16 + col) * SEQ, j, lane);
#pragma unroll
            for (int t = 0; t < QT; ++t) co[t][d] = wmma_f16(va, pb[t], co[t][d]);
        }
    }

    int b = bh >> 2, h = bh & 3;
    int rowoff = hihalf ? 8 : 0;
#pragma unroll
    for (int t = 0; t < QT; ++t) {
        float invl = 1.0f / l[t];
        // ctx[b][s][h*64+d]; O^T element: d per VGPR, query per lane.
        _Float16* bp = ctxh + (b * SEQ + q0 + t * 16 + col) * DIM + h * HD;
#pragma unroll
        for (int d = 0; d < 4; ++d)
#pragma unroll
            for (int r = 0; r < 8; r += 2) {
                int pk = pack_f16(co[t][d][r] * invl, co[t][d][r + 1] * invl);
                *(int*)(bp + d * 16 + r + rowoff) = pk;
            }
    }
}

// ---------------------------------------------------------------------------
// Kernel 4: out = ctx @ Wo + bo (f32), 4 token-tiles x 1 out-tile per wave.
// ---------------------------------------------------------------------------
__global__ void __launch_bounds__(128, 1)
oproj_kernel(const _Float16* __restrict__ ctxh,
             const _Float16* __restrict__ wot,
             const float* __restrict__ bo,
             float* __restrict__ out) {
    int lane = threadIdx.x & 31;
    int wid  = threadIdx.x >> 5;
    int gid  = blockIdx.x * (blockDim.x >> 5) + wid;   // 0..4095
    int nt = gid & 15, ts = gid >> 4;
    int t0 = ts * (16 * TB), n0 = nt * 16;
    int col = lane & 15;

    const _Float16* bcol = wot + (n0 + col) * DIM;
    v8f c[TB];
#pragma unroll
    for (int t = 0; t < TB; ++t) c[t] = {};
#pragma unroll
    for (int cc = 0; cc < 8; ++cc) {
        int cb = cc * 32;
        v16h bf = load_b_frag(bcol, cb, lane);
#pragma unroll
        for (int t = 0; t < TB; ++t) {
            v16h a = load_a_frag(ctxh + (t0 + t * 16 + col) * DIM, cb, lane);
            c[t] = wmma_f16(a, bf, c[t]);
        }
    }
    float bov = bo[n0 + col];
    int rowoff = (lane >> 4) << 3;
#pragma unroll
    for (int t = 0; t < TB; ++t)
#pragma unroll
        for (int r = 0; r < 8; ++r)
            out[(t0 + t * 16 + r + rowoff) * DIM + n0 + col] = c[t][r] + bov;
}

// ---------------------------------------------------------------------------
extern "C" void kernel_launch(void* const* d_in, const int* in_sizes, int n_in,
                              void* d_out, int out_size, void* d_ws, size_t ws_size,
                              hipStream_t stream) {
    const float* x  = (const float*)d_in[0];
    const float* Wq = (const float*)d_in[1];
    const float* bq = (const float*)d_in[2];
    const float* Wk = (const float*)d_in[3];
    const float* bk = (const float*)d_in[4];
    const float* Wv = (const float*)d_in[5];
    const float* bv = (const float*)d_in[6];
    const float* Wo = (const float*)d_in[7];
    const float* bo = (const float*)d_in[8];
    float* out = (float*)d_out;

    // Workspace (f16 elements), ~33 MB; ctx aliases xh (dead after QKV).
    const size_t NX = (size_t)TOK * DIM;     // 4,194,304 halves
    const size_t NW = (size_t)DIM * DIM;     //    65,536 halves
    _Float16* ws   = (_Float16*)d_ws;
    _Float16* xh   = ws;
    _Float16* wqt  = xh + NX;
    _Float16* wkt  = wqt + NW;
    _Float16* wvt  = wkt + NW;
    _Float16* wot  = wvt + NW;
    _Float16* Qh   = wot + NW;
    _Float16* Kh   = Qh + NX;
    _Float16* Vt   = Kh + NX;
    _Float16* ctxh = xh;                     // alias: xh dead after qkv_kernel

    prep_kernel<<<2048, 256, 0, stream>>>(x, Wq, Wk, Wv, Wo, xh, wqt, wkt, wvt, wot);
    // 4096 projection waves (256 token superblocks x 16 out tiles), 4 waves/block
    qkv_kernel<<<1024, 128, 0, stream>>>(xh, wqt, wkt, wvt, bq, bk, bv, Qh, Kh, Vt);
    // 2048 attention waves (32 bh x 64 query superblocks), 4 waves/block
    attn_kernel<<<512, 128, 0, stream>>>(Qh, Kh, Vt, ctxh);
    oproj_kernel<<<1024, 128, 0, stream>>>(ctxh, wot, bo, out);
}
